// TempVAE_34033320853997
// MI455X (gfx1250) — matrix operand order repaired and checked by reading, not compiled, not run on hardware
//
#include <hip/hip_runtime.h>
#include <hip/hip_bf16.h>

typedef __attribute__((ext_vector_type(16))) _Float16 v16h;
typedef __attribute__((ext_vector_type(8)))  _Float16 v8h;
typedef __attribute__((ext_vector_type(8)))  float    v8f;

#define BB 2048
#define TT 512
#define DD 64
#define HH 16
#define LL 10
#define NTILE (BB / 16)
#define LDH 40   // padded f16 row stride for 16..32-wide LDS tiles (80B: conflict-free, 16B aligned)
#define LDI 72   // padded f16 row stride for the 64-wide infz-input tile (144B)
#define LOG2PI_C 1.8378770664093453f

__device__ __forceinline__ int lane_id() { return (int)(threadIdx.x & 31); }

__device__ __forceinline__ v8f wmma16(v16h a, v16h b, v8f c) {
  return __builtin_amdgcn_wmma_f32_16x16x32_f16(false, a, false, b, (short)0, c, false, false);
}

__device__ __forceinline__ float sigm(float x) { return 1.0f / (1.0f + __expf(-x)); }

// B fragment for V_WMMA_F32_16X16X32_F16: output cols [n0,n0+16), K slab [k0,k0+32).
// b[h] = W[n0 + lane%16][k0 + 8*(lane>=16) + (h<8 ? h : h+8)], zero-padded, loads index-clamped.
__device__ v16h make_bfrag(const float* __restrict__ W, int nout, int nin, int n0, int k0) {
  int lane = lane_id();
  int n = n0 + (lane & 15);
  int kb = k0 + ((lane & 16) ? 8 : 0);
  int nc = n < nout ? n : (nout - 1);
  v16h b;
#pragma unroll
  for (int h = 0; h < 16; ++h) {
    int k = kb + (h < 8 ? h : h + 8);
    int kc = k < nin ? k : (nin - 1);
    float w = W[nc * nin + kc];
    b[h] = (_Float16)((n < nout && k < nin) ? w : 0.0f);
  }
  return b;
}

// A fragment from row-major f16 LDS tile [16][ldk]: two 16B ds loads, no conversions.
__device__ __forceinline__ v16h make_afrag_h(const _Float16* a, int ldk, int k0) {
  int lane = lane_id();
  int m = lane & 15;
  int kb = k0 + ((lane & 16) ? 8 : 0);
  const _Float16* row = a + m * ldk + kb;
  v8h lo = *(const v8h*)(row);
  v8h hi = *(const v8h*)(row + 16);
  return __builtin_shufflevector(lo, hi, 0, 1, 2, 3, 4, 5, 6, 7, 8, 9, 10, 11, 12, 13, 14, 15);
}

// A fragment gathered from strided f32 global rows (used by the big GEMM on x).
__device__ __forceinline__ v16h make_afrag_g(const float* base, long stride, int k0) {
  int lane = lane_id();
  int m = lane & 15;
  int kb = k0 + ((lane & 16) ? 8 : 0);
  const float* row = base + (long)m * stride + kb;
  v16h r;
#pragma unroll
  for (int h = 0; h < 8; ++h) {
    r[h]     = (_Float16)row[h];
    r[h + 8] = (_Float16)row[16 + h];
  }
  return r;
}

// Store C/D fragment (rows mb..mb+7, col n) into f16 LDS tile with row stride LDH.
__device__ __forceinline__ void store_ch(_Float16* dst, v8f c) {
  int lane = lane_id();
  int n = lane & 15;
  int mb = (lane & 16) ? 8 : 0;
#pragma unroll
  for (int r = 0; r < 8; ++r) dst[(mb + r) * LDH + n] = (_Float16)c[r];
}

__device__ __forceinline__ void gru_update(v8f* h, const v8f gi[3], const v8f gh[3],
                                           float bi0, float bi1, float bi2,
                                           float bh0, float bh1, float bh2) {
#pragma unroll
  for (int r = 0; r < 8; ++r) {
    float rg = sigm(gi[0][r] + bi0 + gh[0][r] + bh0);
    float zg = sigm(gi[1][r] + bi1 + gh[1][r] + bh1);
    float ng = tanhf(gi[2][r] + bi2 + rg * (gh[2][r] + bh2));
    (*h)[r] = (1.0f - zg) * ng + zg * (*h)[r];
  }
}

// ---- Kernel 1: gi = x @ Wih^T + bih (fwd & bwd), emitted in fragment-native layout ----
// Tile = (t, 16 batch rows). Output halves at [((t*NTILE+bt)*3+j)*256 + n*16 + row].
__global__ void __launch_bounds__(32)
k_gi(const float* __restrict__ x,
     const float* __restrict__ Wf, const float* __restrict__ bf,
     const float* __restrict__ Wb, const float* __restrict__ bb,
     _Float16* __restrict__ gif, _Float16* __restrict__ gib) {
  int bt = (int)(blockIdx.x & (NTILE - 1));
  int t  = (int)(blockIdx.x >> 7);
  long b0 = (long)bt * 16;
  int lane = lane_id();
  int n = lane & 15;
  int mb = (lane & 16) ? 8 : 0;
  v16h bffr[3][2], bfbr[3][2];
#pragma unroll
  for (int j = 0; j < 3; ++j)
#pragma unroll
    for (int s = 0; s < 2; ++s) {
      bffr[j][s] = make_bfrag(Wf, 48, 64, j * 16, 32 * s);
      bfbr[j][s] = make_bfrag(Wb, 48, 64, j * 16, 32 * s);
    }
  const float* base = x + (b0 * TT + t) * DD;
  v16h a0 = make_afrag_g(base, (long)TT * DD, 0);
  v16h a1 = make_afrag_g(base, (long)TT * DD, 32);
#pragma unroll
  for (int j = 0; j < 3; ++j) {
    long fbase = (((long)t * NTILE + bt) * 3 + j) * 256 + n * 16 + mb;
    v8f c = {};
    c = wmma16(a0, bffr[j][0], c);
    c = wmma16(a1, bffr[j][1], c);
    float bias = bf[j * 16 + n];
#pragma unroll
    for (int r = 0; r < 8; ++r) gif[fbase + r] = (_Float16)(c[r] + bias);
    v8f d = {};
    d = wmma16(a0, bfbr[j][0], d);
    d = wmma16(a1, bfbr[j][1], d);
    float bias2 = bb[j * 16 + n];
#pragma unroll
    for (int r = 0; r < 8; ++r) gib[fbase + r] = (_Float16)(d[r] + bias2);
  }
}

// ---------------- Kernel 2: bidirectional GRU scan, writes shifted ctx ----------------
__global__ void __launch_bounds__(32)
k_bigru(const _Float16* __restrict__ gif, const _Float16* __restrict__ gib,
        const float* __restrict__ WhhF, const float* __restrict__ bhhF,
        const float* __restrict__ WhhB, const float* __restrict__ bhhB,
        _Float16* __restrict__ ctx) {
  int dir = (int)(blockIdx.x & 1);
  int bt  = (int)(blockIdx.x >> 1);
  long b0 = (long)bt * 16;
  int lane = lane_id();
  int n = lane & 15;
  int mb = (lane & 16) ? 8 : 0;
  const _Float16* gi = dir ? gib : gif;
  const float* Whh = dir ? WhhB : WhhF;
  const float* bhh = dir ? bhhB : bhhF;
  v16h w0 = make_bfrag(Whh, 48, 16, 0, 0);
  v16h w1 = make_bfrag(Whh, 48, 16, 16, 0);
  v16h w2 = make_bfrag(Whh, 48, 16, 32, 0);
  float bh0 = bhh[n], bh1 = bhh[16 + n], bh2 = bhh[32 + n];
  __shared__ alignas(16) _Float16 hl[16 * LDH];
  for (int i = threadIdx.x; i < 16 * LDH; i += 32) hl[i] = (_Float16)0.0f;
  {  // zero boundary row of ctx (h_right[t=0] / h_left[t=T-1] are zeros)
    int tz = dir ? (TT - 1) : 0;
    int offz = dir ? 16 : 0;
#pragma unroll
    for (int r = 0; r < 8; ++r)
      ctx[((b0 + mb + r) * TT + tz) * 32 + offz + n] = (_Float16)0.0f;
  }
  __syncthreads();
  v8f h = {};
  int off = dir ? 16 : 0;
  for (int tt = 0; tt < TT; ++tt) {
    int t = dir ? (TT - 1 - tt) : tt;
    // hoisted gi loads (fragment-native layout: one 16B load per gate)
    v8h gv[3];
#pragma unroll
    for (int j = 0; j < 3; ++j)
      gv[j] = *(const v8h*)(gi + (((long)t * NTILE + bt) * 3 + j) * 256 + n * 16 + mb);
    v16h a = make_afrag_h(hl, LDH, 0);
    v8f z = {};
    v8f g0 = wmma16(a, w0, z);
    v8f g1 = wmma16(a, w1, z);
    v8f g2 = wmma16(a, w2, z);
#pragma unroll
    for (int r = 0; r < 8; ++r) {
      float rg = sigm((float)gv[0][r] + g0[r] + bh0);
      float zg = sigm((float)gv[1][r] + g1[r] + bh1);
      float ng = tanhf((float)gv[2][r] + rg * (g2[r] + bh2));
      h[r] = (1.0f - zg) * ng + zg * h[r];
    }
    __syncthreads();
    store_ch(hl, h);
    __syncthreads();
    int tw = dir ? (t - 1) : (t + 1);
    if (tw >= 0 && tw < TT) {
#pragma unroll
      for (int r = 0; r < 8; ++r)
        ctx[((b0 + mb + r) * TT + tw) * 32 + off + n] = (_Float16)h[r];
    }
  }
}

// ---------------- Kernel 3: main 512-step scan (3 GRUs + heads + KL/NLL) ----------------
#define NFRAG 42
enum {
  F_IZ_WIH = 0, F_IZ_WHH = 6, F_IN_W1 = 9, F_IN_W2 = 10, F_IN_WMU = 11, F_IN_WCOV = 12,
  F_GZ_WIH = 13, F_GZ_WHH = 16, F_GM_W1 = 19, F_GM_W2 = 20, F_GM_WMU = 21, F_GM_WCOV = 22,
  F_GR_WIH = 23, F_GR_WHH = 26, F_GR_W1 = 29, F_GR_W2 = 30
};

__global__ void __launch_bounds__(32)
k_step(const _Float16* __restrict__ ctx, const float* __restrict__ eps,
       const float* __restrict__ x,
       const float* izWih, const float* izWhh, const float* izbih, const float* izbhh,
       const float* inW1, const float* inb1, const float* inW2, const float* inb2,
       const float* inWmu, const float* inbmu, const float* inWcov, const float* inbcov,
       const float* gzWih, const float* gzWhh, const float* gzbih, const float* gzbhh,
       const float* gmW1, const float* gmb1, const float* gmW2, const float* gmb2,
       const float* gmWmu, const float* gmbmu, const float* gmWcov, const float* gmbcov,
       const float* grWih, const float* grWhh, const float* grbih, const float* grbhh,
       const float* grW1, const float* grb1, const float* grW2, const float* grb2,
       float* __restrict__ part) {
  __shared__ alignas(32) _Float16 wf[NFRAG * 32 * 16];
  __shared__ alignas(16) _Float16 s_in[16 * LDI];   // [z_prev(10) | ctx(32) | zeros]
  __shared__ alignas(16) _Float16 s_z[16 * LDH];    // z (prev, then z_t)
  __shared__ alignas(16) _Float16 s_hz[16 * LDH];
  __shared__ alignas(16) _Float16 s_hgz[16 * LDH];
  __shared__ alignas(16) _Float16 s_hgr[16 * LDH];
  __shared__ alignas(16) _Float16 s_hh[16 * LDH];

  int lane = lane_id();
  int n = lane & 15;
  int mb = (lane & 16) ? 8 : 0;
  long b0 = (long)blockIdx.x * 16;
  v16h* wfv = (v16h*)wf;

  // ---- pack all weight B-fragments into LDS (per-lane contiguous) ----
#pragma unroll
  for (int j = 0; j < 3; ++j)
#pragma unroll
    for (int s = 0; s < 2; ++s)
      wfv[(F_IZ_WIH + j * 2 + s) * 32 + lane] = make_bfrag(izWih, 48, 42, j * 16, 32 * s);
#pragma unroll
  for (int j = 0; j < 3; ++j) wfv[(F_IZ_WHH + j) * 32 + lane] = make_bfrag(izWhh, 48, 16, j * 16, 0);
  wfv[F_IN_W1 * 32 + lane]   = make_bfrag(inW1, 16, 16, 0, 0);
  wfv[F_IN_W2 * 32 + lane]   = make_bfrag(inW2, 16, 16, 0, 0);
  wfv[F_IN_WMU * 32 + lane]  = make_bfrag(inWmu, 10, 16, 0, 0);
  wfv[F_IN_WCOV * 32 + lane] = make_bfrag(inWcov, 10, 16, 0, 0);
#pragma unroll
  for (int j = 0; j < 3; ++j) wfv[(F_GZ_WIH + j) * 32 + lane] = make_bfrag(gzWih, 48, 10, j * 16, 0);
#pragma unroll
  for (int j = 0; j < 3; ++j) wfv[(F_GZ_WHH + j) * 32 + lane] = make_bfrag(gzWhh, 48, 16, j * 16, 0);
  wfv[F_GM_W1 * 32 + lane]   = make_bfrag(gmW1, 16, 16, 0, 0);
  wfv[F_GM_W2 * 32 + lane]   = make_bfrag(gmW2, 16, 16, 0, 0);
  wfv[F_GM_WMU * 32 + lane]  = make_bfrag(gmWmu, 10, 16, 0, 0);
  wfv[F_GM_WCOV * 32 + lane] = make_bfrag(gmWcov, 10, 16, 0, 0);
#pragma unroll
  for (int j = 0; j < 3; ++j) wfv[(F_GR_WIH + j) * 32 + lane] = make_bfrag(grWih, 48, 10, j * 16, 0);
#pragma unroll
  for (int j = 0; j < 3; ++j) wfv[(F_GR_WHH + j) * 32 + lane] = make_bfrag(grWhh, 48, 16, j * 16, 0);
  wfv[F_GR_W1 * 32 + lane] = make_bfrag(grW1, 16, 16, 0, 0);
#pragma unroll
  for (int j = 0; j < 12; ++j) wfv[(F_GR_W2 + j) * 32 + lane] = make_bfrag(grW2, 192, 16, j * 16, 0);

  for (int i = threadIdx.x; i < 16 * LDI; i += 32) s_in[i] = (_Float16)0.0f;
  for (int i = threadIdx.x; i < 16 * LDH; i += 32) {
    s_z[i] = (_Float16)0.0f; s_hz[i] = (_Float16)0.0f; s_hgz[i] = (_Float16)0.0f;
    s_hgr[i] = (_Float16)0.0f; s_hh[i] = (_Float16)0.0f;
  }

  bool valid = (n < LL);
  int nn = valid ? n : 0;
  float izbi0 = izbih[n], izbi1 = izbih[16 + n], izbi2 = izbih[32 + n];
  float izbh0 = izbhh[n], izbh1 = izbhh[16 + n], izbh2 = izbhh[32 + n];
  float gzbi0 = gzbih[n], gzbi1 = gzbih[16 + n], gzbi2 = gzbih[32 + n];
  float gzbh0 = gzbhh[n], gzbh1 = gzbhh[16 + n], gzbh2 = gzbhh[32 + n];
  float grbi0 = grbih[n], grbi1 = grbih[16 + n], grbi2 = grbih[32 + n];
  float grbh0 = grbhh[n], grbh1 = grbhh[16 + n], grbh2 = grbhh[32 + n];
  float b_in1 = inb1[n], b_in2 = inb2[n];
  float b_inmu = valid ? inbmu[nn] : 0.0f;
  float b_incov = valid ? inbcov[nn] : 0.0f;
  float b_gm1 = gmb1[n], b_gm2 = gmb2[n];
  float b_gmmu = valid ? gmbmu[nn] : 0.0f;
  float b_gmcov = valid ? gmbcov[nn] : 0.0f;
  float b_gr1 = grb1[n];
  float b_gr2[12];
#pragma unroll
  for (int j = 0; j < 12; ++j) b_gr2[j] = grb2[j * 16 + n];

  v8f h_z = {}, h_gz = {}, h_gr = {};
  float kl_acc = 0.0f, nll_acc = 0.0f;
  __syncthreads();

  for (int t = 0; t < TT; ++t) {
    // ---- hoisted global loads: ctx -> s_in, x and eps -> registers ----
    {
      int m = n;
      int c0 = (lane & 16) ? 16 : 0;
      const _Float16* cp = ctx + ((b0 + m) * TT + t) * 32 + c0;
#pragma unroll
      for (int i = 0; i < 16; ++i) s_in[m * LDI + 10 + c0 + i] = cp[i];
    }
    float xr[4][8];
#pragma unroll
    for (int j = 0; j < 4; ++j)
#pragma unroll
      for (int r = 0; r < 8; ++r)
        xr[j][r] = x[((b0 + mb + r) * TT + t) * DD + j * 16 + n];
    float er[8];
#pragma unroll
    for (int r = 0; r < 8; ++r) er[r] = eps[((b0 + mb + r) * TT + t) * LL + nn];
    __syncthreads();

    v8f q_mu, q_sig = {}, p_mu, p_sig = {};
    // ---- infz GRU ----
    {
      v16h a0 = make_afrag_h(s_in, LDI, 0);
      v16h a1 = make_afrag_h(s_in, LDI, 32);
      v16h ah = make_afrag_h(s_hz, LDH, 0);
      v8f gi[3], gh[3];
#pragma unroll
      for (int j = 0; j < 3; ++j) {
        v8f c = {};
        c = wmma16(a0, wfv[(F_IZ_WIH + j * 2) * 32 + lane], c);
        c = wmma16(a1, wfv[(F_IZ_WIH + j * 2 + 1) * 32 + lane], c);
        gi[j] = c;
        v8f zz = {};
        gh[j] = wmma16(ah, wfv[(F_IZ_WHH + j) * 32 + lane], zz);
      }
      gru_update(&h_z, gi, gh, izbi0, izbi1, izbi2, izbh0, izbh1, izbh2);
      store_ch(s_hz, h_z);
      __syncthreads();
    }
    // ---- inference head ----
    {
      v16h ah = make_afrag_h(s_hz, LDH, 0);
      v8f zz = {};
      v8f c = wmma16(ah, wfv[F_IN_W1 * 32 + lane], zz);
#pragma unroll
      for (int r = 0; r < 8; ++r) c[r] = fmaxf(c[r] + b_in1, 0.0f);
      store_ch(s_hh, c);
      __syncthreads();
      v16h a2 = make_afrag_h(s_hh, LDH, 0);
      v8f c2 = wmma16(a2, wfv[F_IN_W2 * 32 + lane], zz);
#pragma unroll
      for (int r = 0; r < 8; ++r) c2[r] = fmaxf(c2[r] + b_in2, 0.0f);
      __syncthreads();
      store_ch(s_hh, c2);
      __syncthreads();
      v16h a3 = make_afrag_h(s_hh, LDH, 0);
      q_mu = wmma16(a3, wfv[F_IN_WMU * 32 + lane], zz);
      v8f qc = wmma16(a3, wfv[F_IN_WCOV * 32 + lane], zz);
#pragma unroll
      for (int r = 0; r < 8; ++r) {
        q_mu[r] += b_inmu;
        q_sig[r] = __expf(qc[r] + b_incov);
      }
    }
    // ---- genz GRU (input z_prev) ----
    {
      v16h az = make_afrag_h(s_z, LDH, 0);
      v16h ah = make_afrag_h(s_hgz, LDH, 0);
      v8f gi[3], gh[3], zz = {};
#pragma unroll
      for (int j = 0; j < 3; ++j) {
        gi[j] = wmma16(az, wfv[(F_GZ_WIH + j) * 32 + lane], zz);
        gh[j] = wmma16(ah, wfv[(F_GZ_WHH + j) * 32 + lane], zz);
      }
      gru_update(&h_gz, gi, gh, gzbi0, gzbi1, gzbi2, gzbh0, gzbh1, gzbh2);
      store_ch(s_hgz, h_gz);
      __syncthreads();
    }
    // ---- genz head ----
    {
      v16h ah = make_afrag_h(s_hgz, LDH, 0);
      v8f zz = {};
      v8f c = wmma16(ah, wfv[F_GM_W1 * 32 + lane], zz);
#pragma unroll
      for (int r = 0; r < 8; ++r) c[r] = fmaxf(c[r] + b_gm1, 0.0f);
      __syncthreads();
      store_ch(s_hh, c);
      __syncthreads();
      v16h a2 = make_afrag_h(s_hh, LDH, 0);
      v8f c2 = wmma16(a2, wfv[F_GM_W2 * 32 + lane], zz);
#pragma unroll
      for (int r = 0; r < 8; ++r) c2[r] = fmaxf(c2[r] + b_gm2, 0.0f);
      __syncthreads();
      store_ch(s_hh, c2);
      __syncthreads();
      v16h a3 = make_afrag_h(s_hh, LDH, 0);
      p_mu = wmma16(a3, wfv[F_GM_WMU * 32 + lane], zz);
      v8f pc = wmma16(a3, wfv[F_GM_WCOV * 32 + lane], zz);
#pragma unroll
      for (int r = 0; r < 8; ++r) {
        p_mu[r] += b_gmmu;
        p_sig[r] = __expf(pc[r] + b_gmcov);
      }
    }
    // ---- KL + reparameterized z_t ----
    {
      v8f kl;
#pragma unroll
      for (int r = 0; r < 8; ++r) {
        float qs = q_sig[r], ps = p_sig[r];
        float dm = q_mu[r] - p_mu[r];
        float term = __logf(ps / qs) + (qs * qs + dm * dm) / (2.0f * ps * ps) - 0.5f;
        kl[r] = valid ? term : 0.0f;
      }
#pragma unroll
      for (int mask = 1; mask < 16; mask <<= 1)
#pragma unroll
        for (int r = 0; r < 8; ++r) kl[r] += __shfl_xor(kl[r], mask);
      if ((lane & 15) == 0) {
#pragma unroll
        for (int r = 0; r < 8; ++r) kl_acc += kl[r];
      }
#pragma unroll
      for (int r = 0; r < 8; ++r) {
        float zt = valid ? (q_mu[r] + q_sig[r] * er[r]) : 0.0f;
        s_z[(mb + r) * LDH + n] = (_Float16)zt;
        if (valid) s_in[(mb + r) * LDI + n] = (_Float16)zt;  // z_prev for next step
      }
      __syncthreads();
    }
    // ---- genr GRU (input z_t) ----
    {
      v16h az = make_afrag_h(s_z, LDH, 0);
      v16h ah = make_afrag_h(s_hgr, LDH, 0);
      v8f gi[3], gh[3], zz = {};
#pragma unroll
      for (int j = 0; j < 3; ++j) {
        gi[j] = wmma16(az, wfv[(F_GR_WIH + j) * 32 + lane], zz);
        gh[j] = wmma16(ah, wfv[(F_GR_WHH + j) * 32 + lane], zz);
      }
      gru_update(&h_gr, gi, gh, grbi0, grbi1, grbi2, grbh0, grbh1, grbh2);
      store_ch(s_hgr, h_gr);
      __syncthreads();
    }
    // ---- genr readout + NLL ----
    {
      v16h ah = make_afrag_h(s_hgr, LDH, 0);
      v8f zz = {};
      v8f c = wmma16(ah, wfv[F_GR_W1 * 32 + lane], zz);
#pragma unroll
      for (int r = 0; r < 8; ++r) c[r] = fmaxf(c[r] + b_gr1, 0.0f);
      __syncthreads();
      store_ch(s_hh, c);
      __syncthreads();
      v16h a2 = make_afrag_h(s_hh, LDH, 0);
      v8f s1 = {}, s2 = {}, s3 = {}, s4 = {};
#pragma unroll
      for (int j = 0; j < 4; ++j) {
        v8f rmu = wmma16(a2, wfv[(F_GR_W2 + j) * 32 + lane], zz);
        v8f rdg = wmma16(a2, wfv[(F_GR_W2 + 4 + j) * 32 + lane], zz);
        v8f rw  = wmma16(a2, wfv[(F_GR_W2 + 8 + j) * 32 + lane], zz);
        float bm = b_gr2[j], bd = b_gr2[4 + j], bw = b_gr2[8 + j];
#pragma unroll
        for (int r = 0; r < 8; ++r) {
          float mu = rmu[r] + bm;
          float dg = __expf(rdg[r] + bd) + 1e-4f;
          float ww = rw[r] + bw;
          float dinv = 1.0f / dg;
          float delta = xr[j][r] - mu;
          float wd = ww * dinv;
          s1[r] += delta * delta * dinv;
          s2[r] += wd * delta;
          s3[r] += wd * ww;
          s4[r] += __logf(dg);
        }
      }
#pragma unroll
      for (int mask = 1; mask < 16; mask <<= 1)
#pragma unroll
        for (int r = 0; r < 8; ++r) {
          s1[r] += __shfl_xor(s1[r], mask);
          s2[r] += __shfl_xor(s2[r], mask);
          s3[r] += __shfl_xor(s3[r], mask);
          s4[r] += __shfl_xor(s4[r], mask);
        }
      if ((lane & 15) == 0) {
#pragma unroll
        for (int r = 0; r < 8; ++r) {
          float cap = 1.0f + s3[r];
          float maha = s1[r] - s2[r] * s2[r] / cap;
          float logdet = s4[r] + __logf(cap);
          float logp = -0.5f * (64.0f * LOG2PI_C + logdet + maha);
          nll_acc -= logp;
        }
      }
    }
  }  // t loop

  kl_acc += __shfl_xor(kl_acc, 16);
  nll_acc += __shfl_xor(nll_acc, 16);
  if (lane == 0) {
    part[blockIdx.x * 2 + 0] = nll_acc;
    part[blockIdx.x * 2 + 1] = kl_acc;
  }
}

// ---------------- Kernel 4: deterministic final reduction ----------------
__global__ void k_fin(const float* __restrict__ part, float* __restrict__ out) {
  if (blockIdx.x == 0 && threadIdx.x == 0) {
    float nll = 0.0f, kl = 0.0f;
    for (int i = 0; i < NTILE; ++i) {
      nll += part[i * 2 + 0];
      kl  += part[i * 2 + 1];
    }
    out[0] = nll / (float)BB;  // mean over batch, summed over time
    out[1] = kl / (float)BB;
  }
}

extern "C" void kernel_launch(void* const* d_in, const int* in_sizes, int n_in,
                              void* d_out, int out_size, void* d_ws, size_t ws_size,
                              hipStream_t stream) {
  const float* x      = (const float*)d_in[0];
  const float* eps    = (const float*)d_in[1];
  const float* biWihF = (const float*)d_in[2];
  const float* biWhhF = (const float*)d_in[3];
  const float* bibihF = (const float*)d_in[4];
  const float* bibhhF = (const float*)d_in[5];
  const float* biWihB = (const float*)d_in[6];
  const float* biWhhB = (const float*)d_in[7];
  const float* bibihB = (const float*)d_in[8];
  const float* bibhhB = (const float*)d_in[9];
  const float* izWih  = (const float*)d_in[10];
  const float* izWhh  = (const float*)d_in[11];
  const float* izbih  = (const float*)d_in[12];
  const float* izbhh  = (const float*)d_in[13];
  const float* inW1   = (const float*)d_in[14];
  const float* inb1   = (const float*)d_in[15];
  const float* inW2   = (const float*)d_in[16];
  const float* inb2   = (const float*)d_in[17];
  const float* inWmu  = (const float*)d_in[18];
  const float* inbmu  = (const float*)d_in[19];
  const float* inWcov = (const float*)d_in[20];
  const float* inbcov = (const float*)d_in[21];
  const float* gzWih  = (const float*)d_in[22];
  const float* gzWhh  = (const float*)d_in[23];
  const float* gzbih  = (const float*)d_in[24];
  const float* gzbhh  = (const float*)d_in[25];
  const float* gmW1   = (const float*)d_in[26];
  const float* gmb1   = (const float*)d_in[27];
  const float* gmW2   = (const float*)d_in[28];
  const float* gmb2   = (const float*)d_in[29];
  const float* gmWmu  = (const float*)d_in[30];
  const float* gmbmu  = (const float*)d_in[31];
  const float* gmWcov = (const float*)d_in[32];
  const float* gmbcov = (const float*)d_in[33];
  const float* grWih  = (const float*)d_in[34];
  const float* grWhh  = (const float*)d_in[35];
  const float* grbih  = (const float*)d_in[36];
  const float* grbhh  = (const float*)d_in[37];
  const float* grW1   = (const float*)d_in[38];
  const float* grb1   = (const float*)d_in[39];
  const float* grW2   = (const float*)d_in[40];
  const float* grb2   = (const float*)d_in[41];

  char* ws = (char*)d_ws;
  const size_t GI_BYTES  = (size_t)BB * TT * 48 * 2;  // f16 gate preactivations (fragment layout)
  const size_t CTX_BYTES = (size_t)BB * TT * 32 * 2;  // f16 shifted context
  _Float16* gif  = (_Float16*)(ws);
  _Float16* gib  = (_Float16*)(ws + GI_BYTES);
  _Float16* ctxp = (_Float16*)(ws + 2 * GI_BYTES);
  float*    part = (float*)(ws + 2 * GI_BYTES + CTX_BYTES);

  // Phase 1: gi = x @ Wih^T + bih (both directions) — one big streaming WMMA GEMM
  k_gi<<<dim3(TT * NTILE), dim3(32), 0, stream>>>(x, biWihF, bibihF, biWihB, bibihB, gif, gib);
  // Phase 2: forward + backward GRU recurrences, write time-shifted ctx
  k_bigru<<<dim3(2 * NTILE), dim3(32), 0, stream>>>(gif, gib, biWhhF, bibhhF, biWhhB, bibhhB, ctxp);
  // Phase 3: main sequential scan (42 WMMAs/step per 16-row tile)
  k_step<<<dim3(NTILE), dim3(32), 0, stream>>>(
      ctxp, eps, x,
      izWih, izWhh, izbih, izbhh,
      inW1, inb1, inW2, inb2, inWmu, inbmu, inWcov, inbcov,
      gzWih, gzWhh, gzbih, gzbhh,
      gmW1, gmb1, gmW2, gmb2, gmWmu, gmbmu, gmWcov, gmbcov,
      grWih, grWhh, grbih, grbhh,
      grW1, grb1, grW2, grb2,
      part);
  // Phase 4: deterministic reduction into d_out = {nll_sum / B, kl_sum / B}
  k_fin<<<dim3(1), dim3(32), 0, stream>>>(part, (float*)d_out);
  (void)in_sizes; (void)n_in; (void)out_size; (void)ws_size;
}
